// M3Site_43258910605423
// MI455X (gfx1250) — matrix operand order repaired and checked by reading, not compiled, not run on hardware
//
#include <hip/hip_runtime.h>
#include <hip/hip_bf16.h>

typedef __bf16 bf16_t;
typedef __attribute__((ext_vector_type(16))) __bf16 v16bf;
typedef __attribute__((ext_vector_type(8)))  float  v8f;

#if defined(__has_builtin)
# if __has_builtin(__builtin_amdgcn_global_load_async_to_lds_b128)
#  define HAVE_ASYNC_LDS 1
# endif
# if __has_builtin(__builtin_amdgcn_s_wait_asynccnt)
#  define HAVE_WAIT_ASYNC_BUILTIN 1
# endif
#endif

#if defined(HAVE_ASYNC_LDS)
typedef int v4i_t __attribute__((vector_size(16)));
typedef __attribute__((address_space(1))) v4i_t* as1_v4i;
typedef __attribute__((address_space(3))) v4i_t* as3_v4i;
#endif

__device__ __forceinline__ void wait_async0() {
#if defined(HAVE_ASYNC_LDS)
# if defined(HAVE_WAIT_ASYNC_BUILTIN)
  __builtin_amdgcn_s_wait_asynccnt(0);
# else
  asm volatile("s_wait_asynccnt 0x0" ::: "memory");
# endif
#endif
}

#define NG     16
#define LMAXD  1024
#define NODESD 512
#define EDIM   1536
#define SDIM   256
#define NROWS  (NG * LMAXD)     /* 16384 dense rows */
#define NVALID (NG * NODESD)    /* 8192 valid rows  */
#define HDIM   256
#define FFDIM  128
#define CONDD  768
#define NCLS   7
#define DTOT   1792             /* EDIM + SDIM */

#define BM  128
#define BN  64
#define BK  32
#define LDK 40                  /* padded LDS row (bf16 elems) */

// ---------------------------------------------------------------- fragments
__device__ __forceinline__ v16bf ld_fragA(const bf16_t* prow, int khi) {
  union { v16bf v; uint4 q[2]; } u;
  u.q[0] = *(const uint4*)(prow + khi);
  u.q[1] = *(const uint4*)(prow + 16 + khi);
  return u.v;
}
__device__ __forceinline__ v16bf ld_fragB(const bf16_t* prow, int kbase) {
  union { v16bf v; uint4 q[2]; } u;
  u.q[0] = *(const uint4*)(prow + kbase);
  u.q[1] = *(const uint4*)(prow + 16 + kbase);
  return u.v;
}

// ---------------------------------------------------------------- WMMA GEMM
// C[M,N] = act( alpha * A[M,K](bf16) @ B[N,K](bf16)^T + bias + Res )
// Outputs: C (f32, optional), Cb (bf16 row-major, optional),
//          CbT (bf16 transposed [N,M], optional).
// gbias=1: bias is per-graph [16][N], graph = row>>10.
// Double-buffered pipeline; tiles pulled into LDS with async-load-to-LDS
// (ASYNCcnt) when the toolchain exposes the gfx1250 builtin.
__global__ __launch_bounds__(256) void k_gemm(
    const bf16_t* __restrict__ A, long sAg, int lda,
    const bf16_t* __restrict__ B, long sBg, int ldb,
    float*  __restrict__ C,   long sCg,   int ldc,
    bf16_t* __restrict__ Cb,  long sCbg,  int ldcb,
    bf16_t* __restrict__ CbT, long sCbTg, int ldcbt,
    const float* __restrict__ bias, int gbias,
    const float* __restrict__ Res, long sRg, int ldr,
    float alpha, int relu_f, int N, int K)
{
  __shared__ __align__(16) bf16_t As[2][BM * LDK];
  __shared__ __align__(16) bf16_t Bs[2][BN * LDK];
  const int gz = blockIdx.z;
  A += (long)gz * sAg;  B += (long)gz * sBg;
  if (C)   C   += (long)gz * sCg;
  if (Cb)  Cb  += (long)gz * sCbg;
  if (CbT) CbT += (long)gz * sCbTg;
  if (Res) Res += (long)gz * sRg;

  const int m0 = blockIdx.y * BM, n0 = blockIdx.x * BN;
  const int tid = threadIdx.x, lane = tid & 31, wv = tid >> 5;
  const int wm = (wv & 3) * 32, wn = (wv >> 2) * 32;
  const int fr    = lane & 15;
  const int khi   = (lane >> 4) << 3;   // A frag K select
  const int kbase = (lane >> 4) << 3;   // B frag K select

  v8f zero = {0.f,0.f,0.f,0.f,0.f,0.f,0.f,0.f};
  v8f acc[2][2] = {{zero, zero}, {zero, zero}};

  const int ar = tid >> 1, ac = (tid & 1) * 16;   // A tile: 128 rows x 32
  const int br = tid >> 2, bc = (tid & 3) * 8;    // B tile:  64 rows x 32
  const bf16_t* aSrc = A + (long)(m0 + ar) * lda + ac;
  const bf16_t* bSrc = B + (long)(n0 + br) * ldb + bc;

  auto issue_tile = [&](int buf, int k0) {
#if defined(HAVE_ASYNC_LDS)
    __builtin_amdgcn_global_load_async_to_lds_b128(
        (as1_v4i)(void*)(aSrc + k0),
        (as3_v4i)(void*)&As[buf][ar * LDK + ac], 0, 0);
    __builtin_amdgcn_global_load_async_to_lds_b128(
        (as1_v4i)(void*)(aSrc + k0 + 8),
        (as3_v4i)(void*)&As[buf][ar * LDK + ac + 8], 0, 0);
    __builtin_amdgcn_global_load_async_to_lds_b128(
        (as1_v4i)(void*)(bSrc + k0),
        (as3_v4i)(void*)&Bs[buf][br * LDK + bc], 0, 0);
#else
    *(uint4*)&As[buf][ar * LDK + ac]     = *(const uint4*)(aSrc + k0);
    *(uint4*)&As[buf][ar * LDK + ac + 8] = *(const uint4*)(aSrc + k0 + 8);
    *(uint4*)&Bs[buf][br * LDK + bc]     = *(const uint4*)(bSrc + k0);
#endif
  };

  issue_tile(0, 0);
  int buf = 0;
  for (int k0 = 0; k0 < K; k0 += BK, buf ^= 1) {
    wait_async0();
    __syncthreads();
    if (k0 + BK < K) issue_tile(buf ^ 1, k0 + BK);
    if (k0 + 2 * BK < K) __builtin_prefetch(aSrc + k0 + 2 * BK, 0, 1);

    v16bf af0 = ld_fragA(&As[buf][(wm +      fr) * LDK], khi);
    v16bf af1 = ld_fragA(&As[buf][(wm + 16 + fr) * LDK], khi);
    v16bf bf0 = ld_fragB(&Bs[buf][(wn +      fr) * LDK], kbase);
    v16bf bf1 = ld_fragB(&Bs[buf][(wn + 16 + fr) * LDK], kbase);

    acc[0][0] = __builtin_amdgcn_wmma_f32_16x16x32_bf16(false, af0, false, bf0, (short)0, acc[0][0], false, false);
    acc[0][1] = __builtin_amdgcn_wmma_f32_16x16x32_bf16(false, af0, false, bf1, (short)0, acc[0][1], false, false);
    acc[1][0] = __builtin_amdgcn_wmma_f32_16x16x32_bf16(false, af1, false, bf0, (short)0, acc[1][0], false, false);
    acc[1][1] = __builtin_amdgcn_wmma_f32_16x16x32_bf16(false, af1, false, bf1, (short)0, acc[1][1], false, false);
  }

  const int mlo = (lane >> 4) << 3;
  const int nn  = lane & 15;
  #pragma unroll
  for (int i = 0; i < 2; ++i)
  #pragma unroll
  for (int j = 0; j < 2; ++j) {
    #pragma unroll
    for (int r = 0; r < 8; ++r) {
      int row = m0 + wm + i * 16 + mlo + r;
      int col = n0 + wn + j * 16 + nn;
      float v = acc[i][j][r] * alpha;
      if (bias)  v += gbias ? bias[(row >> 10) * N + col] : bias[col];
      if (Res)   v += Res[(long)row * ldr + col];
      if (relu_f) v = fmaxf(v, 0.f);
      if (C)   C[(long)row * ldc + col] = v;
      if (Cb)  Cb[(long)row * ldcb + col] = (bf16_t)v;
      if (CbT) CbT[(long)col * ldcbt + row] = (bf16_t)v;
    }
  }
}

// ---------------------------------------------------------------- helpers
__global__ void k_scatter(const float* __restrict__ esm, const float* __restrict__ egnn,
                          float* __restrict__ esm_d, float* __restrict__ egnn_d,
                          bf16_t* __restrict__ esm_b, bf16_t* __restrict__ egnn_b)
{
  const int row = blockIdx.x;           // dense row 0..16383
  const int g = row >> 10, p = row & 1023;
  const bool valid = p < NODESD;
  const long src = (long)g * NODESD + p;
  for (int j = threadIdx.x; j < EDIM; j += blockDim.x) {
    float v = valid ? esm[src * EDIM + j] : 0.f;
    esm_d[(long)row * EDIM + j] = v;
    esm_b[(long)row * EDIM + j] = (bf16_t)v;
  }
  for (int j = threadIdx.x; j < SDIM; j += blockDim.x) {
    float v = valid ? egnn[src * SDIM + j] : 0.f;
    egnn_d[(long)row * SDIM + j] = v;
    egnn_b[(long)row * SDIM + j] = (bf16_t)v;
  }
}

// W:[K,N] f32 row-major -> out:[N,K] bf16 (transposed for coalesced GEMM B path)
__global__ void k_convT(const float* __restrict__ W, bf16_t* __restrict__ out, int K, int N)
{
  long i = (long)blockIdx.x * 256 + threadIdx.x;
  if (i < (long)K * N) {
    int k = (int)(i / N), n = (int)(i % N);
    out[(long)n * K + k] = (bf16_t)W[i];
  }
}

// condbias[g][f] = bb1[f] + sum_c func[g][c] * w1[(d1+c)][f]
__global__ void k_condbias(const float* __restrict__ func, const float* __restrict__ w1,
                           const float* __restrict__ bb1, float* __restrict__ out, int d1)
{
  const int g = blockIdx.x, f = threadIdx.x;
  const float* wr = w1 + (long)d1 * FFDIM;
  float s = bb1[f];
  for (int c = 0; c < CONDD; ++c) s += func[g * CONDD + c] * wr[(long)c * FFDIM + f];
  out[g * FFDIM + f] = s;
}

__global__ __launch_bounds__(256) void k_softmax(const float* __restrict__ S, bf16_t* __restrict__ P)
{
  __shared__ float red[256];
  const long row = blockIdx.x;
  const float* s = S + row * (long)LMAXD;
  const int t = threadIdx.x;
  float mx = -3.0e38f;
  #pragma unroll
  for (int k = 0; k < 4; ++k) mx = fmaxf(mx, s[t + k * 256]);
  red[t] = mx; __syncthreads();
  for (int o = 128; o > 0; o >>= 1) { if (t < o) red[t] = fmaxf(red[t], red[t + o]); __syncthreads(); }
  mx = red[0]; __syncthreads();
  float e[4], sum = 0.f;
  #pragma unroll
  for (int k = 0; k < 4; ++k) { e[k] = __expf(s[t + k * 256] - mx); sum += e[k]; }
  red[t] = sum; __syncthreads();
  for (int o = 128; o > 0; o >>= 1) { if (t < o) red[t] += red[t + o]; __syncthreads(); }
  const float inv = 1.f / red[0];
  bf16_t* p = P + row * (long)LMAXD;
  #pragma unroll
  for (int k = 0; k < 4; ++k) p[t + k * 256] = (bf16_t)(e[k] * inv);
}

__global__ __launch_bounds__(256) void k_ln(float* __restrict__ X, int ld, int d,
    const float* __restrict__ gam, const float* __restrict__ bet,
    bf16_t* __restrict__ Xb, int ldb)
{
  __shared__ float red[256];
  const long row = blockIdx.x;
  float* x = X + row * (long)ld;
  const int t = threadIdx.x;
  float s = 0.f;
  for (int j = t; j < d; j += 256) s += x[j];
  red[t] = s; __syncthreads();
  for (int o = 128; o > 0; o >>= 1) { if (t < o) red[t] += red[t + o]; __syncthreads(); }
  const float mean = red[0] / d; __syncthreads();
  float v = 0.f;
  for (int j = t; j < d; j += 256) { float q = x[j] - mean; v += q * q; }
  red[t] = v; __syncthreads();
  for (int o = 128; o > 0; o >>= 1) { if (t < o) red[t] += red[t + o]; __syncthreads(); }
  const float inv = rsqrtf(red[0] / d + 1e-5f);
  bf16_t* xb = Xb ? Xb + row * (long)ldb : (bf16_t*)0;
  for (int j = t; j < d; j += 256) {
    float o_ = (x[j] - mean) * inv * gam[j] + bet[j];
    x[j] = o_;
    if (xb) xb[j] = (bf16_t)o_;
  }
}

__global__ __launch_bounds__(256) void k_gate(const float* __restrict__ fusion,
    const float* __restrict__ esm_d, const float* __restrict__ egnn_d,
    const float* __restrict__ wfcw, const float* __restrict__ wfcb, float* __restrict__ gate)
{
  __shared__ float red[256];
  const long row = blockIdx.x;
  const int t = threadIdx.x;
  float s = 0.f;
  for (int j = t; j < DTOT; j += 256) s += fusion[row * DTOT + j] * wfcw[j];
  for (int j = t; j < EDIM; j += 256) s += esm_d[row * EDIM + j] * wfcw[DTOT + j];
  for (int j = t; j < SDIM; j += 256) s += egnn_d[row * SDIM + j] * wfcw[DTOT + EDIM + j];
  red[t] = s; __syncthreads();
  for (int o = 128; o > 0; o >>= 1) { if (t < o) red[t] += red[t + o]; __syncthreads(); }
  if (t == 0) gate[row] = 1.f / (1.f + __expf(-(red[0] + wfcb[0])));
}

__global__ void k_mix(const float* __restrict__ fusion, const float* __restrict__ esm_d,
                      const float* __restrict__ egnn_d, const float* __restrict__ gate,
                      bf16_t* __restrict__ out)
{
  const long row = blockIdx.x;
  const float w = gate[row];
  for (int j = threadIdx.x; j < DTOT; j += 256) {
    float tot = (j < EDIM) ? esm_d[row * EDIM + j] : egnn_d[row * SDIM + (j - EDIM)];
    out[row * DTOT + j] = (bf16_t)(w * fusion[row * DTOT + j] + (1.f - w) * tot);
  }
}

__global__ __launch_bounds__(256) void k_bnstats(const float* __restrict__ X, float* __restrict__ mv)
{
  __shared__ float r1[256], r2[256];
  const int ch = blockIdx.x, t = threadIdx.x;
  float s = 0.f, s2 = 0.f;
  for (int r = t; r < NROWS; r += 256) { float v = X[(long)r * HDIM + ch]; s += v; s2 += v * v; }
  r1[t] = s; r2[t] = s2; __syncthreads();
  for (int o = 128; o > 0; o >>= 1) { if (t < o) { r1[t] += r1[t + o]; r2[t] += r2[t + o]; } __syncthreads(); }
  if (t == 0) { float m = r1[0] / NROWS; mv[ch] = m; mv[HDIM + ch] = r2[0] / NROWS - m * m; }
}

__global__ void k_bnapply(float* __restrict__ X, const float* __restrict__ mv,
                          const float* __restrict__ g, const float* __restrict__ b)
{
  const long i = (long)blockIdx.x * 256 + threadIdx.x;
  const int ch = (int)(i & (HDIM - 1));
  float o = (X[i] - mv[ch]) * rsqrtf(mv[HDIM + ch] + 1e-5f) * g[ch] + b[ch];
  X[i] = fmaxf(o, 0.f);
}

__global__ void k_zero(float* p, int n) { int i = threadIdx.x; if (i < n) p[i] = 0.f; }

// fc2 + softmax logits + center-loss accumulation for valid rows only
__global__ __launch_bounds__(64) void k_head(const float* __restrict__ X,
    const float* __restrict__ fc2w, const float* __restrict__ fc2b,
    const float* __restrict__ centers, const int* __restrict__ y, float* __restrict__ out)
{
  __shared__ float logits[NCLS];
  const int vrow = blockIdx.x;               // 0..8191
  const int g = vrow >> 9, p = vrow & 511;
  const long drow = (long)g * LMAXD + p;
  if (threadIdx.x < NCLS) {
    float s = fc2b[threadIdx.x];
    for (int k = 0; k < HDIM; ++k) s += X[drow * HDIM + k] * fc2w[k * NCLS + threadIdx.x];
    logits[threadIdx.x] = s;
  }
  __syncthreads();
  if (threadIdx.x == 0) {
    float mx = -3.0e38f;
    for (int c = 0; c < NCLS; ++c) mx = fmaxf(mx, logits[c]);
    float e[NCLS], sum = 0.f;
    for (int c = 0; c < NCLS; ++c) { e[c] = __expf(logits[c] - mx); sum += e[c]; }
    const float inv = 1.f / sum;
    for (int c = 0; c < NCLS; ++c) out[(long)vrow * NCLS + c] = e[c] * inv;
    const float* ctr = centers + y[vrow] * NCLS;
    float cl = 0.f;
    for (int c = 0; c < NCLS; ++c) { float d = logits[c] - ctr[c]; cl += d * d; }
    atomicAdd(&out[(long)NVALID * NCLS], 0.5f * cl / (float)NVALID);
  }
}

__global__ void k_inter(const float* __restrict__ centers, float* __restrict__ out)
{
  __shared__ float red[32];
  const int t = threadIdx.x;
  float s = 0.f;
  if (t < 21) {                               // upper-triangular pairs of 7 centers
    int i = 0, j = 0, c = t;
    for (i = 0; i < NCLS; ++i) { int cnt = NCLS - 1 - i; if (c < cnt) { j = i + 1 + c; break; } c -= cnt; }
    float d2 = 1e-12f;
    for (int k = 0; k < NCLS; ++k) { float d = centers[i * NCLS + k] - centers[j * NCLS + k]; d2 += d * d; }
    s = fmaxf(0.1f - sqrtf(d2), 0.f);
  }
  red[t] = s; __syncthreads();
  for (int o = 16; o > 0; o >>= 1) { if (t < o) red[t] += red[t + o]; __syncthreads(); }
  if (t == 0) out[(long)NVALID * NCLS + 1] = red[0] / 21.f;
}

// ---------------------------------------------------------------- launch
extern "C" void kernel_launch(void* const* d_in, const int* in_sizes, int n_in,
                              void* d_out, int out_size, void* d_ws, size_t ws_size,
                              hipStream_t stream)
{
  (void)in_sizes; (void)n_in; (void)out_size; (void)ws_size;
  const float* esm  = (const float*)d_in[0];
  const float* egnn = (const float*)d_in[1];
  const float* func = (const float*)d_in[2];
  const int*   yv   = (const int*)d_in[5];
  // p1
  const float *kw1=(const float*)d_in[6],  *kb1=(const float*)d_in[7];
  const float *vw1=(const float*)d_in[8],  *vb1=(const float*)d_in[9];
  const float *ow1=(const float*)d_in[10], *ob1=(const float*)d_in[11];
  const float *g1_1=(const float*)d_in[12],*b1_1=(const float*)d_in[13];
  const float *w1_1=(const float*)d_in[14],*bb1_1=(const float*)d_in[15];
  const float *w2_1=(const float*)d_in[16],*bb2_1=(const float*)d_in[17];
  const float *g2_1=(const float*)d_in[18],*b2_1=(const float*)d_in[19];
  // p2
  const float *kw2=(const float*)d_in[20], *kb2=(const float*)d_in[21];
  const float *vw2=(const float*)d_in[22], *vb2=(const float*)d_in[23];
  const float *ow2=(const float*)d_in[24], *ob2=(const float*)d_in[25];
  const float *g1_2=(const float*)d_in[26],*b1_2=(const float*)d_in[27];
  const float *w1_2=(const float*)d_in[28],*bb1_2=(const float*)d_in[29];
  const float *w2_2=(const float*)d_in[30],*bb2_2=(const float*)d_in[31];
  const float *g2_2=(const float*)d_in[32],*b2_2=(const float*)d_in[33];
  // ph
  const float *wfcw=(const float*)d_in[34],*wfcb=(const float*)d_in[35];
  const float *fc1w=(const float*)d_in[36],*fc1b=(const float*)d_in[37];
  const float *bng=(const float*)d_in[38], *bnb=(const float*)d_in[39];
  const float *fc2w=(const float*)d_in[40],*fc2b=(const float*)d_in[41];
  const float *centers=(const float*)d_in[42];
  float* out = (float*)d_out;

  char* wp = (char*)d_ws;
  auto alloc = [&](size_t bytes) -> char* {
    char* p = wp; wp += (bytes + 255) & ~(size_t)255; return p;
  };
  float*  ESM_D  = (float*) alloc((size_t)NROWS*EDIM*4);
  float*  EGNN_D = (float*) alloc((size_t)NROWS*SDIM*4);
  bf16_t* ESM_B  = (bf16_t*)alloc((size_t)NROWS*EDIM*2);
  bf16_t* EGNN_B = (bf16_t*)alloc((size_t)NROWS*SDIM*2);
  bf16_t* KP_B   = (bf16_t*)alloc((size_t)NROWS*EDIM*2);
  bf16_t* VPT_B  = (bf16_t*)alloc((size_t)EDIM*NROWS*2);   // V projection, TRANSPOSED [d][NROWS]
  float*  SCORES = (float*) alloc((size_t)NG*LMAXD*LMAXD*4);
  bf16_t* PROBS_B= (bf16_t*)alloc((size_t)NG*LMAXD*LMAXD*2);
  bf16_t* CTX_B  = (bf16_t*)alloc((size_t)NROWS*EDIM*2);
  float*  QLN    = (float*) alloc((size_t)NROWS*EDIM*4);
  bf16_t* QLN_B  = (bf16_t*)alloc((size_t)NROWS*EDIM*2);
  bf16_t* H_B    = (bf16_t*)alloc((size_t)NROWS*FFDIM*2);
  float*  FUSION = (float*) alloc((size_t)NROWS*DTOT*4);
  bf16_t* MIX_B  = (bf16_t*)alloc((size_t)NROWS*DTOT*2);
  float*  FC1O   = (float*) alloc((size_t)NROWS*HDIM*4);
  float*  GATE   = (float*) alloc((size_t)NROWS*4);
  float*  CONDB1 = (float*) alloc((size_t)NG*FFDIM*4);
  float*  CONDB2 = (float*) alloc((size_t)NG*FFDIM*4);
  float*  MEANVAR= (float*) alloc((size_t)2*HDIM*4);
  bf16_t* KW1T   = (bf16_t*)alloc((size_t)SDIM*EDIM*2);
  bf16_t* VW1T   = (bf16_t*)alloc((size_t)SDIM*EDIM*2);
  bf16_t* OW1T   = (bf16_t*)alloc((size_t)SDIM*SDIM*2);
  bf16_t* W1A1T  = (bf16_t*)alloc((size_t)FFDIM*SDIM*2);
  bf16_t* W21T   = (bf16_t*)alloc((size_t)SDIM*FFDIM*2);
  bf16_t* KW2T   = (bf16_t*)alloc((size_t)EDIM*SDIM*2);
  bf16_t* VW2T   = (bf16_t*)alloc((size_t)EDIM*SDIM*2);
  bf16_t* OW2T   = (bf16_t*)alloc((size_t)EDIM*EDIM*2);
  bf16_t* W1A2T  = (bf16_t*)alloc((size_t)FFDIM*EDIM*2);
  bf16_t* W22T   = (bf16_t*)alloc((size_t)EDIM*FFDIM*2);
  bf16_t* FC1T   = (bf16_t*)alloc((size_t)HDIM*DTOT*2);

  auto gemm = [&](const bf16_t* A, long sAg, int lda,
                  const bf16_t* B, long sBg, int ldb,
                  float* C, long sCg, int ldc, bf16_t* Cb, long sCbg, int ldcb,
                  bf16_t* CbT, long sCbTg, int ldcbt,
                  const float* bias, int gbias, const float* Res, long sRg, int ldr,
                  float alpha, int relu_f, int M, int N, int K, int Z) {
    dim3 grid(N / BN, M / BM, Z);
    k_gemm<<<grid, dim3(256), 0, stream>>>(A, sAg, lda, B, sBg, ldb,
        C, sCg, ldc, Cb, sCbg, ldcb, CbT, sCbTg, ldcbt,
        bias, gbias, Res, sRg, ldr, alpha, relu_f, N, K);
  };
  auto convT = [&](const float* W, bf16_t* T, int K, int N) {
    long tot = (long)K * N;
    k_convT<<<(int)((tot + 255) / 256), 256, 0, stream>>>(W, T, K, N);
  };

  // ---- dense scatter + weight conversion (transpose to [N,K] bf16)
  k_scatter<<<NROWS, 256, 0, stream>>>(esm, egnn, ESM_D, EGNN_D, ESM_B, EGNN_B);
  convT(kw1, KW1T, EDIM, SDIM);   convT(vw1, VW1T, EDIM, SDIM);
  convT(ow1, OW1T, SDIM, SDIM);   convT(w1_1, W1A1T, SDIM, FFDIM);
  convT(w2_1, W21T, FFDIM, SDIM);
  convT(kw2, KW2T, SDIM, EDIM);   convT(vw2, VW2T, SDIM, EDIM);
  convT(ow2, OW2T, EDIM, EDIM);   convT(w1_2, W1A2T, EDIM, FFDIM);
  convT(w2_2, W22T, FFDIM, EDIM); convT(fc1w, FC1T, DTOT, HDIM);
  k_condbias<<<NG, FFDIM, 0, stream>>>(func, w1_1, bb1_1, CONDB1, SDIM);
  k_condbias<<<NG, FFDIM, 0, stream>>>(func, w1_2, bb1_2, CONDB2, EDIM);

  const long SGA = (long)LMAXD;  // per-graph row blocks

  // ---- s1: Q=egnn (d=256), K=V=esm
  gemm(ESM_B,0,EDIM, KW1T,0,EDIM, 0,0,0, KP_B,0,SDIM, 0,0,0, kb1,0, 0,0,0, 1.f,0, NROWS,SDIM,EDIM, 1);
  gemm(ESM_B,0,EDIM, VW1T,0,EDIM, 0,0,0, 0,0,0, VPT_B,0,NROWS, vb1,0, 0,0,0, 1.f,0, NROWS,SDIM,EDIM, 1);
  gemm(EGNN_B,SGA*SDIM,SDIM, KP_B,SGA*SDIM,SDIM, SCORES,SGA*LMAXD,LMAXD, 0,0,0, 0,0,0,
       0,0, 0,0,0, 0.0625f,0, LMAXD,LMAXD,SDIM, NG);
  k_softmax<<<NROWS, 256, 0, stream>>>(SCORES, PROBS_B);
  gemm(PROBS_B,SGA*LMAXD,LMAXD, VPT_B,SGA,NROWS, 0,0,0, CTX_B,SGA*SDIM,SDIM, 0,0,0,
       0,0, 0,0,0, 1.f,0, LMAXD,SDIM,LMAXD, NG);
  gemm(CTX_B,0,SDIM, OW1T,0,SDIM, QLN,0,SDIM, 0,0,0, 0,0,0, ob1,0, EGNN_D,0,SDIM, 1.f,0, NROWS,SDIM,SDIM, 1);
  k_ln<<<NROWS, 256, 0, stream>>>(QLN, SDIM, SDIM, g1_1, b1_1, QLN_B, SDIM);
  gemm(QLN_B,0,SDIM, W1A1T,0,SDIM, 0,0,0, H_B,0,FFDIM, 0,0,0, CONDB1,1, 0,0,0, 1.f,1, NROWS,FFDIM,SDIM, 1);
  gemm(H_B,0,FFDIM, W21T,0,FFDIM, FUSION,0,DTOT, 0,0,0, 0,0,0, bb2_1,0, QLN,0,SDIM, 1.f,0, NROWS,SDIM,FFDIM, 1);
  k_ln<<<NROWS, 256, 0, stream>>>(FUSION, DTOT, SDIM, g2_1, b2_1, (bf16_t*)0, 0);

  // ---- s2: Q=esm (d=1536), K=V=egnn
  gemm(EGNN_B,0,SDIM, KW2T,0,SDIM, 0,0,0, KP_B,0,EDIM, 0,0,0, kb2,0, 0,0,0, 1.f,0, NROWS,EDIM,SDIM, 1);
  gemm(EGNN_B,0,SDIM, VW2T,0,SDIM, 0,0,0, 0,0,0, VPT_B,0,NROWS, vb2,0, 0,0,0, 1.f,0, NROWS,EDIM,SDIM, 1);
  gemm(ESM_B,SGA*EDIM,EDIM, KP_B,SGA*EDIM,EDIM, SCORES,SGA*LMAXD,LMAXD, 0,0,0, 0,0,0,
       0,0, 0,0,0, 0.025515518f,0, LMAXD,LMAXD,EDIM, NG);
  k_softmax<<<NROWS, 256, 0, stream>>>(SCORES, PROBS_B);
  gemm(PROBS_B,SGA*LMAXD,LMAXD, VPT_B,SGA,NROWS, 0,0,0, CTX_B,SGA*EDIM,EDIM, 0,0,0,
       0,0, 0,0,0, 1.f,0, LMAXD,EDIM,LMAXD, NG);
  gemm(CTX_B,0,EDIM, OW2T,0,EDIM, QLN,0,EDIM, 0,0,0, 0,0,0, ob2,0, ESM_D,0,EDIM, 1.f,0, NROWS,EDIM,EDIM, 1);
  k_ln<<<NROWS, 256, 0, stream>>>(QLN, EDIM, EDIM, g1_2, b1_2, QLN_B, EDIM);
  gemm(QLN_B,0,EDIM, W1A2T,0,EDIM, 0,0,0, H_B,0,FFDIM, 0,0,0, CONDB2,1, 0,0,0, 1.f,1, NROWS,FFDIM,EDIM, 1);
  gemm(H_B,0,FFDIM, W22T,0,FFDIM, FUSION+SDIM,0,DTOT, 0,0,0, 0,0,0, bb2_2,0, QLN,0,EDIM, 1.f,0, NROWS,EDIM,FFDIM, 1);
  k_ln<<<NROWS, 256, 0, stream>>>(FUSION + SDIM, DTOT, EDIM, g2_2, b2_2, (bf16_t*)0, 0);

  // ---- head: gate, mix, fc1, BN(train)+relu, fc2 + losses
  k_gate<<<NROWS, 256, 0, stream>>>(FUSION, ESM_D, EGNN_D, wfcw, wfcb, GATE);
  k_mix<<<NROWS, 256, 0, stream>>>(FUSION, ESM_D, EGNN_D, GATE, MIX_B);
  gemm(MIX_B,0,DTOT, FC1T,0,DTOT, FC1O,0,HDIM, 0,0,0, 0,0,0, fc1b,0, 0,0,0, 1.f,0, NROWS,HDIM,DTOT, 1);
  k_bnstats<<<HDIM, 256, 0, stream>>>(FC1O, MEANVAR);
  k_bnapply<<<(NROWS * HDIM) / 256, 256, 0, stream>>>(FC1O, MEANVAR, bng, bnb);
  k_zero<<<1, 32, 0, stream>>>(out + (size_t)NVALID * NCLS, 2);
  k_head<<<NVALID, 64, 0, stream>>>(FC1O, fc2w, fc2b, centers, yv, out);
  k_inter<<<1, 32, 0, stream>>>(centers, out);
}
